// Attention_77403900609148
// MI455X (gfx1250) — matrix operand order repaired and checked by reading, compiled, not run on hardware
//
#include <hip/hip_runtime.h>
#include <math.h>

#define BB 64
#define SS 2048
#define DD 512
#define HH 512

typedef __attribute__((ext_vector_type(16))) __bf16 v16bf;
typedef __attribute__((ext_vector_type(8)))  __bf16 v8bf;
typedef __attribute__((ext_vector_type(4)))  __bf16 v4bf;
typedef __attribute__((ext_vector_type(8)))  float  v8f;

// ---- A loader: W_ctx pre-converted to bf16 row-major [H][D] --------------
// A operand k order chosen here (matches B): lane m<16 -> M=m, K=0..15;
// lane>=16 -> M=m, K=16..31. Caller folds half*16 into p; we load
// p[0..7] and p[8..15] as two 16-byte bf16 loads (no conversions).
__device__ __forceinline__ v16bf loadA_bf(const __bf16* __restrict__ p) {
    v8bf lo = *reinterpret_cast<const v8bf*>(p);
    v8bf hi = *reinterpret_cast<const v8bf*>(p + 8);
    v16bf r;
#pragma unroll
    for (int i = 0; i < 8; ++i) { r[i] = lo[i]; r[i + 8] = hi[i]; }
    return r;
}

// ---- B loader: streaming context f32 -> bf16 in registers ---------------
// B operand k order chosen here (matches A): lane n -> N=n%16; lanes 0-15 hold K=0..15,
// lanes 16-31 hold K=16..31 (caller folds half*16 into p). 16 consecutive f32.
__device__ __forceinline__ v16bf loadB16(const float* __restrict__ p) {
    const float4* q = reinterpret_cast<const float4*>(p);
    float4 b0 = q[0], b1 = q[1], b2 = q[2], b3 = q[3];
    v16bf r;
    r[0]  = (__bf16)b0.x; r[1]  = (__bf16)b0.y; r[2]  = (__bf16)b0.z; r[3]  = (__bf16)b0.w;
    r[4]  = (__bf16)b1.x; r[5]  = (__bf16)b1.y; r[6]  = (__bf16)b1.z; r[7]  = (__bf16)b1.w;
    r[8]  = (__bf16)b2.x; r[9]  = (__bf16)b2.y; r[10] = (__bf16)b2.z; r[11] = (__bf16)b2.w;
    r[12] = (__bf16)b3.x; r[13] = (__bf16)b3.y; r[14] = (__bf16)b3.z; r[15] = (__bf16)b3.w;
    return r;
}

// C/D 16x16 f32 layout: VGPR r, lane c: M = r + 8*(c>=16), N = c%16.
__device__ __forceinline__ float tile_att(const v8f& acc, int h0, int half,
                                          const float* __restrict__ b_ctx,
                                          const float* __restrict__ inp_b,
                                          const float* __restrict__ V) {
    float s = 0.f;
#pragma unroll
    for (int r = 0; r < 8; ++r) {
        int h = h0 + r + 8 * half;
        float t = acc[r] + b_ctx[h] + inp_b[h];
        s += V[h] * tanhf(t);
    }
    return s;
}

// ---- kernel 0: pre-convert W_ctx f32 -> bf16 (reused by all blocks) ------
__global__ void __launch_bounds__(256) wctx_bf16_kernel(const float* __restrict__ W_ctx,
                                                        __bf16* __restrict__ Wbf) {
    int i = (blockIdx.x * 256 + threadIdx.x) * 4;       // [0, H*D) step 4
    float4 a = *reinterpret_cast<const float4*>(W_ctx + i);
    v4bf o;
    o[0] = (__bf16)a.x; o[1] = (__bf16)a.y; o[2] = (__bf16)a.z; o[3] = (__bf16)a.w;
    *reinterpret_cast<v4bf*>(Wbf + i) = o;
}

// ---- kernel 1: inp = input @ W_in^T + b_in -------------------------------
__global__ void __launch_bounds__(256) inp_kernel(const float* __restrict__ input,
                                                  const float* __restrict__ W_in,
                                                  const float* __restrict__ b_in,
                                                  float* __restrict__ inp) {
    int idx = blockIdx.x * 256 + threadIdx.x;           // [0, B*H)
    int b = idx / HH, h = idx % HH;
    const float4* x = reinterpret_cast<const float4*>(input + (size_t)b * DD);
    const float4* w = reinterpret_cast<const float4*>(W_in + (size_t)h * DD);
    float s = 0.f;
#pragma unroll 4
    for (int j = 0; j < DD / 4; ++j) {
        float4 a = x[j], c = w[j];
        s += a.x * c.x + a.y * c.y + a.z * c.z + a.w * c.w;
    }
    inp[idx] = s + b_in[h];
}

// ---- kernel 2: WMMA attention energies -----------------------------------
// grid = (S/32, B); block = 256 = 8 waves. Wave w owns H rows [w*64,w*64+64)
// as four M-tiles; block covers 32 s columns as two N-tiles. Each A tile is
// reused by both N-tiles: 8 WMMAs per K-step of 32.
__global__ void __launch_bounds__(256) att_wmma_kernel(const float* __restrict__ context,
                                                       const __bf16* __restrict__ Wbf,
                                                       const float* __restrict__ b_ctx,
                                                       const float* __restrict__ V,
                                                       const float* __restrict__ inp,
                                                       float* __restrict__ att) {
    const int stile = blockIdx.x;            // 0..S/32-1
    const int b     = blockIdx.y;            // 0..B-1
    const int tid   = threadIdx.x;
    const int wave  = tid >> 5;              // 0..7
    const int lane  = tid & 31;
    const int half  = lane >> 4;             // 0 or 1
    const int nm    = lane & 15;             // row (A) / col (B,C) index
    const int s0    = stile * 32;

    __shared__ float satt[8][32];

    // two B-tile bases: columns s0+nm and s0+16+nm, K-half offset half*16
    const float* ctxRow0 = context + ((size_t)(b * SS + s0 + nm)) * DD + half * 16;
    const float* ctxRow1 = ctxRow0 + (size_t)16 * DD;
    // bf16 A base: row h = wave*64+nm, K-run offset half*16
    const __bf16* Ab = Wbf + (size_t)(wave * 64 + nm) * DD + half * 16;

    v8f a00 = {}, a10 = {}, a20 = {}, a30 = {};   // N-tile 0
    v8f a01 = {}, a11 = {}, a21 = {}, a31 = {};   // N-tile 1

#pragma unroll 2
    for (int d0 = 0; d0 < DD; d0 += 32) {
        v16bf B0 = loadB16(ctxRow0 + d0);
        v16bf B1 = loadB16(ctxRow1 + d0);
        v16bf A0 = loadA_bf(Ab + d0);
        v16bf A1 = loadA_bf(Ab + 16 * DD + d0);
        v16bf A2 = loadA_bf(Ab + 32 * DD + d0);
        v16bf A3 = loadA_bf(Ab + 48 * DD + d0);
        a00 = __builtin_amdgcn_wmma_f32_16x16x32_bf16(false, A0, false, B0, (short)0, a00, false, false);
        a01 = __builtin_amdgcn_wmma_f32_16x16x32_bf16(false, A0, false, B1, (short)0, a01, false, false);
        a10 = __builtin_amdgcn_wmma_f32_16x16x32_bf16(false, A1, false, B0, (short)0, a10, false, false);
        a11 = __builtin_amdgcn_wmma_f32_16x16x32_bf16(false, A1, false, B1, (short)0, a11, false, false);
        a20 = __builtin_amdgcn_wmma_f32_16x16x32_bf16(false, A2, false, B0, (short)0, a20, false, false);
        a21 = __builtin_amdgcn_wmma_f32_16x16x32_bf16(false, A2, false, B1, (short)0, a21, false, false);
        a30 = __builtin_amdgcn_wmma_f32_16x16x32_bf16(false, A3, false, B0, (short)0, a30, false, false);
        a31 = __builtin_amdgcn_wmma_f32_16x16x32_bf16(false, A3, false, B1, (short)0, a31, false, false);
    }

    const float* inp_b = inp + (size_t)b * HH;
    const int h0w = wave * 64;
    float attp0 = 0.f, attp1 = 0.f;
    attp0 += tile_att(a00, h0w +  0, half, b_ctx, inp_b, V);
    attp0 += tile_att(a10, h0w + 16, half, b_ctx, inp_b, V);
    attp0 += tile_att(a20, h0w + 32, half, b_ctx, inp_b, V);
    attp0 += tile_att(a30, h0w + 48, half, b_ctx, inp_b, V);
    attp1 += tile_att(a01, h0w +  0, half, b_ctx, inp_b, V);
    attp1 += tile_att(a11, h0w + 16, half, b_ctx, inp_b, V);
    attp1 += tile_att(a21, h0w + 32, half, b_ctx, inp_b, V);
    attp1 += tile_att(a31, h0w + 48, half, b_ctx, inp_b, V);

    // fold lane halves (same N column; M offset by 8 already summed in-lane)
    attp0 += __shfl_xor(attp0, 16, 32);
    attp1 += __shfl_xor(attp1, 16, 32);
    if (lane < 16) { satt[wave][lane] = attp0; satt[wave][lane + 16] = attp1; }
    __syncthreads();
    if (tid < 32) {
        float t = 0.f;
#pragma unroll
        for (int w = 0; w < 8; ++w) t += satt[w][tid];
        att[(size_t)b * SS + s0 + tid] = t;
    }
}

// ---- kernel 3a: zero scratch accumulators --------------------------------
__global__ void __launch_bounds__(256) init_kernel(float* __restrict__ cbar, int* __restrict__ cnt) {
    int i = blockIdx.x * 256 + threadIdx.x;
    if (i < BB * DD) cbar[i] = 0.f;
    if (i == 0) *cnt = 0;
}

// ---- kernel 3b: global popcount of mask (branch condition) ---------------
__global__ void __launch_bounds__(256) mask_count_kernel(const int* __restrict__ mask,
                                                         int* __restrict__ cnt) {
    __shared__ int sc[256];
    int i = blockIdx.x * 256 + threadIdx.x;
    sc[threadIdx.x] = (mask[i] != 0) ? 1 : 0;
    __syncthreads();
    for (int off = 128; off > 0; off >>= 1) {
        if (threadIdx.x < off) sc[threadIdx.x] += sc[threadIdx.x + off];
        __syncthreads();
    }
    if (threadIdx.x == 0) atomicAdd(cnt, sc[0]);
}

// ---- kernel 4: mask branch + softmax over S, one block per batch ---------
__global__ void __launch_bounds__(256) softmax_kernel(const float* __restrict__ att,
                                                      const float* __restrict__ alpha_mask,
                                                      const int* __restrict__ mask,
                                                      const int* __restrict__ cnt,
                                                      float* __restrict__ alpha) {
    const int b = blockIdx.x;
    const int t = threadIdx.x;
    __shared__ float red[256];
    const int c = *cnt;
    const bool infb = (c > 0) && (c == SS);

    float a[SS / 256];
    float mx = -INFINITY;
#pragma unroll
    for (int j = 0; j < SS / 256; ++j) {
        int s = t + j * 256;
        float v = att[(size_t)b * SS + s];
        if (infb) v = (mask[(size_t)b * SS + s] != 0) ? -INFINITY : v;
        else      v *= alpha_mask[(size_t)b * SS + s];
        a[j] = v;
        mx = fmaxf(mx, v);
    }
    red[t] = mx; __syncthreads();
    for (int off = 128; off > 0; off >>= 1) {
        if (t < off) red[t] = fmaxf(red[t], red[t + off]);
        __syncthreads();
    }
    mx = red[0]; __syncthreads();

    float sum = 0.f;
#pragma unroll
    for (int j = 0; j < SS / 256; ++j) { a[j] = expf(a[j] - mx); sum += a[j]; }
    red[t] = sum; __syncthreads();
    for (int off = 128; off > 0; off >>= 1) {
        if (t < off) red[t] += red[t + off];
        __syncthreads();
    }
    const float inv = 1.0f / red[0];
#pragma unroll
    for (int j = 0; j < SS / 256; ++j)
        alpha[(size_t)b * SS + t + j * 256] = a[j] * inv;
}

// ---- kernel 5: c_bar[b,d] = sum_s alpha[b,s] * context[b,s,d] ------------
__global__ void __launch_bounds__(256) cbar_kernel(const float* __restrict__ context,
                                                   const float* __restrict__ alpha,
                                                   float* __restrict__ cbar) {
    const int b = blockIdx.y;
    const int chunk = blockIdx.x;           // 0..7
    const int t = threadIdx.x;              // covers d = t and t+256
    float acc0 = 0.f, acc1 = 0.f;
    const int sbeg = chunk * (SS / 8);
    for (int s = sbeg; s < sbeg + SS / 8; ++s) {
        float a = alpha[(size_t)b * SS + s];
        const float* row = context + ((size_t)(b * SS + s)) * DD;
        acc0 += a * row[t];
        acc1 += a * row[t + 256];
    }
    atomicAdd(&cbar[(size_t)b * DD + t], acc0);
    atomicAdd(&cbar[(size_t)b * DD + t + 256], acc1);
}

// ---- kernel 6: hidden = W_ctx @ c_bar + b_ctx ----------------------------
__global__ void __launch_bounds__(256) hidden_kernel(const float* __restrict__ W_ctx,
                                                     const float* __restrict__ b_ctx,
                                                     const float* __restrict__ cbar,
                                                     float* __restrict__ hidden) {
    int idx = blockIdx.x * 256 + threadIdx.x;           // [0, B*H)
    int b = idx / HH, h = idx % HH;
    const float4* x = reinterpret_cast<const float4*>(cbar + (size_t)b * DD);
    const float4* w = reinterpret_cast<const float4*>(W_ctx + (size_t)h * DD);
    float s = 0.f;
#pragma unroll 4
    for (int j = 0; j < DD / 4; ++j) {
        float4 a = x[j], c = w[j];
        s += a.x * c.x + a.y * c.y + a.z * c.z + a.w * c.w;
    }
    hidden[idx] = s + b_ctx[h];
}

extern "C" void kernel_launch(void* const* d_in, const int* in_sizes, int n_in,
                              void* d_out, int out_size, void* d_ws, size_t ws_size,
                              hipStream_t stream) {
    const float* input      = (const float*)d_in[0];   // [B,D]
    const float* context    = (const float*)d_in[1];   // [B,S,D]
    const float* alpha_mask = (const float*)d_in[2];   // [B,S]
    const float* W_in       = (const float*)d_in[3];   // [H,D]
    const float* b_in       = (const float*)d_in[4];   // [H]
    const float* W_ctx      = (const float*)d_in[5];   // [H,D]
    const float* b_ctx      = (const float*)d_in[6];   // [H]
    const float* V          = (const float*)d_in[7];   // [H]
    const int*   mask       = (const int*)d_in[8];     // [B,S]

    float* out_hidden = (float*)d_out;                 // [B,H]
    float* out_alpha  = out_hidden + BB * HH;          // [B,S]

    float* ws_att  = (float*)d_ws;                     // [B,S]
    float* ws_inp  = ws_att + BB * SS;                 // [B,H]
    float* ws_cbar = ws_inp + BB * HH;                 // [B,D]
    int*   ws_cnt  = (int*)(ws_cbar + BB * DD);        // [1] (+pad)
    __bf16* ws_wbf = (__bf16*)(ws_cnt + 4);            // [H*D] bf16

    wctx_bf16_kernel<<<(HH * DD / 4) / 256, 256, 0, stream>>>(W_ctx, ws_wbf);
    init_kernel<<<(BB * DD) / 256, 256, 0, stream>>>(ws_cbar, ws_cnt);
    mask_count_kernel<<<(BB * SS) / 256, 256, 0, stream>>>(mask, ws_cnt);
    inp_kernel<<<(BB * HH) / 256, 256, 0, stream>>>(input, W_in, b_in, ws_inp);
    att_wmma_kernel<<<dim3(SS / 32, BB), 256, 0, stream>>>(context, ws_wbf, b_ctx, V, ws_inp, ws_att);
    softmax_kernel<<<BB, 256, 0, stream>>>(ws_att, alpha_mask, mask, ws_cnt, out_alpha);
    cbar_kernel<<<dim3(8, BB), 256, 0, stream>>>(context, out_alpha, ws_cbar);
    hidden_kernel<<<(BB * HH) / 256, 256, 0, stream>>>(W_ctx, b_ctx, ws_cbar, out_hidden);
}
